// ImageTransitionPredictionModel_58935541236522
// MI455X (gfx1250) — compile-verified, matrix operand order checked by reading, not run on hardware
//
#include <hip/hip_runtime.h>

// ---------------- problem constants ----------------
#define BATCH 16384
#define ADIM  8
#define EDIM  128
#define HDIM  128
#define GDIM  512          // 4*H
#define TILE_B 16

typedef __attribute__((ext_vector_type(16))) __bf16 v16bf;
typedef __attribute__((ext_vector_type(8)))  float  v8f;
typedef __attribute__((ext_vector_type(4)))  unsigned v4u;
typedef __attribute__((ext_vector_type(8)))  int      v8i;
typedef __attribute__((ext_vector_type(4)))  int      v4i;

__device__ __forceinline__ v8f wmma_bf16(v16bf a, v16bf b, v8f c) {
  return __builtin_amdgcn_wmma_f32_16x16x32_bf16(false, a, false, b, (short)0, c, false, false);
}

__device__ __forceinline__ float sigmoid_f(float x) { return 1.f / (1.f + __expf(-x)); }

// ---- TDM 1-D tile load: global -> LDS, 8-byte elements (nbytes % 8 == 0) ----
// D# per CDNA5 ISA; groups 2/3 zero (<=2-D tensor).
// This toolchain exposes the 6-arg builtin: (g0, g1, g2, g3, g4, cpol).
__device__ __forceinline__ void tdm_load_1d(unsigned lds_addr, const void* gptr, unsigned nbytes) {
  unsigned ne = nbytes >> 3;                          // elements of 8 B
  unsigned long long ga = (unsigned long long)gptr;
  v4u g0;
  g0[0] = 1u;                                         // count=1, user descriptor
  g0[1] = lds_addr;                                   // LDS byte address
  g0[2] = (unsigned)ga;                               // global_addr[31:0]
  g0[3] = (unsigned)((ga >> 32) & 0x01FFFFFFu) | 0x80000000u;  // addr[56:32] | type=2
  v8i g1;
  g1[0] = (int)(3u << 16);                            // wg_mask=0, data_size=3 (8B)
  g1[1] = (int)((ne & 0xFFFFu) << 16);                // tensor_dim0[15:0] @ bits[63:48]
  g1[2] = (int)((ne >> 16) & 0xFFFFu) | (1 << 16);    // tensor_dim0[31:16], tensor_dim1=1
  g1[3] = (int)((ne & 0xFFFFu) << 16);                // tensor_dim1 hi=0, tile_dim0=ne
  g1[4] = 1;                                          // tile_dim1=1, tile_dim2=0
  g1[5] = (int)ne;                                    // tensor_dim0_stride[31:0]
  g1[6] = 0;
  g1[7] = 0;
  v4i z4 = {0, 0, 0, 0};
  v8i z8 = {0, 0, 0, 0, 0, 0, 0, 0};
  __builtin_amdgcn_tensor_load_to_lds(g0, g1, z4, z4, z8, 0);
}
__device__ __forceinline__ unsigned lds_addr_of(const void* p) {
  return (unsigned)(unsigned long long)p;             // generic ptr low 32 bits == LDS addr
}

// ---------------- workspace layout (bytes) ----------------
#define WS_XP    0                    // f32  [B][512]            33554432 B
#define WS_HSB   33554432             // bf16 [B][128]             4194304 B
#define WS_EWB   37748736             // bf16 [25][128][64]         409600 B
#define WS_WIHB  38158336             // bf16 [5][512][32] zero-pad 163840 B
#define WS_C2WB  38322176             // bf16 [64][288] K'-order     36864 B
#define WS_T1WB  38359040             // bf16 [32][576] K'-order     36864 B
#define WS_DWB   38395904             // bf16 [1600][128]           409600 B

// ============ one-time weight prep: permute + f32->bf16 ============
__global__ __launch_bounds__(256) void prep_weights_kernel(
    const float* __restrict__ ew, const float* __restrict__ wih,
    const float* __restrict__ c2w, const float* __restrict__ t1w,
    const float* __restrict__ dw,
    __bf16* __restrict__ ewb, __bf16* __restrict__ wihb,
    __bf16* __restrict__ c2wb, __bf16* __restrict__ t1wb,
    __bf16* __restrict__ dwb)
{
  int idx = blockIdx.x * 256 + threadIdx.x;          // grid covers 204800
  if (idx < 204800) {
    int kk = idx / 8192, rem = idx % 8192, n = rem >> 6, kloc = rem & 63;
    ewb[idx] = (__bf16)ew[n * 1600 + kk * 64 + kloc];     // [kk][n][64]
    dwb[idx] = (__bf16)dw[idx];                            // [n][k] as-is
  }
  if (idx < 81920) {
    int kk = idx / 16384, rem = idx % 16384, n = rem >> 5, kloc = rem & 31;
    int k = kk * 32 + kloc;
    wihb[idx] = (k < 136) ? (__bf16)wih[n * 136 + k] : (__bf16)0.f;   // [kk][n][32]
  }
  if (idx < 18432) {
    {   // conv2: [o][k'] with k' = (dy*3+dx)*32 + ci
      int o = idx / 288, kp = idx % 288, e = kp >> 5, ci = kp & 31;
      c2wb[idx] = (__bf16)c2w[o * 288 + ci * 9 + e];
    }
    {   // deconv1 effective conv: [o][k'] with k' = (dy*3+dx)*64 + ci, flipped+swapped
      int o = idx / 576, kp = idx % 576, e = kp >> 6, ci = kp & 63;
      int dy = e / 3, dx = e % 3;
      t1wb[idx] = (__bf16)t1w[ci * 288 + o * 9 + (2 - dy) * 3 + (2 - dx)];
    }
  }
}

// ===================== encoder =====================
#define ENC_OFF_H1 0        // bf16 [7*7][16][32] conv1 out planes, zero-padded (50176 B)
#define ENC_OFF_H2 50176    // bf16 [16][1600] conv2 out in flatten order       (51200 B)
#define ENC_OFF_R  101376   // TDM-owned: w2(36864) / bk double-buffer 2x16384 / red f32(16384)
#define ENC_OFF_Z  138240   // f32: img padded (9408) early, then z [16][160] (10240 B)
#define ENC_OFF_W1 148480   // f32 conv1 w(864) + b(32)                         (3584 B)
#define ENC_LDS    152064

__global__ __launch_bounds__(512) void encoder_kernel(
    const float* __restrict__ img, const float* __restrict__ action,
    const float* __restrict__ c1w, const float* __restrict__ c1b,
    const __bf16* __restrict__ c2wb, const float* __restrict__ c2b,
    const __bf16* __restrict__ ewb,  const float* __restrict__ eb,
    const __bf16* __restrict__ wihb, const float* __restrict__ bih,
    const float* __restrict__ bhh, float* __restrict__ xp)
{
  extern __shared__ char smem[];
  __bf16* s_h1  = (__bf16*)(smem + ENC_OFF_H1);   // [plane(49)][b(16)][ci(32)]
  __bf16* s_h2  = (__bf16*)(smem + ENC_OFF_H2);   // [b][1600]
  __bf16* s_w2  = (__bf16*)(smem + ENC_OFF_R);    // [o(64)][k'(288)]
  __bf16* s_bk  = (__bf16*)(smem + ENC_OFF_R);    // [n][kchunk], double-buffered
  float*  s_red = (float*) (smem + ENC_OFF_R);    // [2][16][128] split-K partials
  float*  s_img = (float*) (smem + ENC_OFF_Z);    // padded img (early lifetime)
  float*  s_z   = (float*) (smem + ENC_OFF_Z);    // z||action (late lifetime)
  float*  s_w1  = (float*) (smem + ENC_OFF_W1);
  float*  s_b1  = s_w1 + 864;

  const int tid  = threadIdx.x;
  const int lane = tid & 31;
  const int wave = tid >> 5;
  const int b0   = blockIdx.x * TILE_B;
  const int row  = lane & 15;
  const int khalf = (lane & 16) ? 16 : 0;
  const int mofs  = (lane & 16) ? 8 : 0;

  // kick off conv2-weight DMA into R while we do conv1 (R untouched until then)
  if (wave == 0) tdm_load_1d(lds_addr_of(s_w2), c2wb, 36864);

  // zero padded h1 planes, zero padded img, stage conv1 weights
  {
    unsigned* z1 = (unsigned*)s_h1;
    for (int i = tid; i < 12544; i += 512) z1[i] = 0u;          // 25088 bf16
    for (int i = tid; i < 2352;  i += 512) s_img[i] = 0.f;
    for (int i = tid; i < 896;   i += 512) {
      if (i < 864) s_w1[i] = c1w[i]; else s_b1[i - 864] = c1b[i - 864];
    }
  }
  __syncthreads();
  // stage img tile interior into padded layout [b][c][7][7]
  for (int idx = tid; idx < TILE_B * 75; idx += 512) {
    int b = idx / 75, rem = idx % 75, c = rem / 25, p = rem % 25;
    s_img[(b * 3 + c) * 49 + (p / 5 + 1) * 7 + (p % 5 + 1)] = img[(size_t)(b0 + b) * 75 + rem];
  }
  __syncthreads();
  // conv1 + relu -> plane-major bf16 h1: s_h1[(y*7+x)*512 + b*32 + c]
  for (int idx = tid; idx < TILE_B * 32 * 25; idx += 512) {
    int b = idx / 800, rem = idx % 800, c = rem / 25, p = rem % 25;
    int py = p / 5, px = p % 5;
    float acc = s_b1[c];
#pragma unroll
    for (int ci = 0; ci < 3; ++ci)
#pragma unroll
      for (int dy = 0; dy < 3; ++dy)
#pragma unroll
        for (int dx = 0; dx < 3; ++dx)
          acc += s_img[(b * 3 + ci) * 49 + (py + dy) * 7 + (px + dx)] *
                 s_w1[c * 27 + ci * 9 + dy * 3 + dx];
    s_h1[((py + 1) * 7 + (px + 1)) * 512 + b * 32 + c] = (__bf16)fmaxf(acc, 0.f);
  }
  if (wave == 0) __builtin_amdgcn_s_wait_tensorcnt(0);   // conv2 weights landed
  __syncthreads();
  // conv2 via WMMA implicit GEMM: n-tile fixed per wave (weight reuse), positions strided
  {
    const int nt = wave & 3, pg = wave >> 2;
    const int n = nt * 16 + row;
    const __bf16* wrow = s_w2 + n * 288 + khalf;
    const float bias = c2b[n];
    for (int p = pg; p < 25; p += 4) {
      int py = p / 5, px = p % 5;
      const __bf16* abase = s_h1 + (py * 7 + px) * 512 + row * 32 + khalf;
      v8f acc = {};
#pragma unroll
      for (int kk = 0; kk < 9; ++kk) {
        const int dy = kk / 3, dx = kk % 3;
        v16bf a = *(const v16bf*)(abase + (dy * 7 + dx) * 512);
        v16bf b = *(const v16bf*)(wrow + kk * 32);
        acc = wmma_bf16(a, b, acc);
      }
#pragma unroll
      for (int r = 0; r < 8; ++r) {
        float v = fmaxf(acc[r] + bias, 0.f);
        s_h2[(r + mofs) * 1600 + n * 25 + p] = (__bf16)v;   // flatten order c*25+p
      }
    }
  }
  __syncthreads();                       // all waves done reading s_w2 (R region)
  // enc linear: 64-wide K chunks, TDM double-buffered, all 16 waves split-K
  v8f accz = {};
  {
    const int nt = wave & 7, ksel = wave >> 3;
    if (wave == 0) tdm_load_1d(lds_addr_of(s_bk), ewb, 16384);     // chunk 0 -> buf 0
    for (int kk2 = 0; kk2 < 25; ++kk2) {
      if (wave == 0) __builtin_amdgcn_s_wait_tensorcnt(0);
      __syncthreads();
      if (wave == 0 && kk2 < 24)
        tdm_load_1d(lds_addr_of(s_bk) + ((kk2 + 1) & 1) * 16384,
                    ewb + (kk2 + 1) * 8192, 16384);                // prefetch next chunk
      const __bf16* bkbuf = s_bk + (kk2 & 1) * 8192;
      v16bf a = *(const v16bf*)(s_h2 + row * 1600 + kk2 * 64 + ksel * 32 + khalf);
      v16bf b = *(const v16bf*)(bkbuf + (nt * 16 + row) * 64 + ksel * 32 + khalf);
      accz = wmma_bf16(a, b, accz);
    }
    __syncthreads();
#pragma unroll
    for (int r = 0; r < 8; ++r)
      s_red[ksel * 2048 + (r + mofs) * 128 + nt * 16 + row] = accz[r];
    for (int idx = tid; idx < TILE_B * 24; idx += 512) {
      int b = idx / 24, j = idx % 24;
      s_z[b * 160 + 136 + j] = 0.f;
    }
    __syncthreads();
    for (int idx = tid; idx < 2048; idx += 512) {
      int b = idx >> 7, n = idx & 127;
      s_z[b * 160 + n] = s_red[idx] + s_red[2048 + idx] + eb[n];
    }
    for (int idx = tid; idx < TILE_B * ADIM; idx += 512) {
      int b = idx / ADIM, j = idx % ADIM;
      s_z[b * 160 + 128 + j] = action[(size_t)(b0 + b) * ADIM + j];
    }
  }
  // xp = x(16x136) @ wih^T + (bih+bhh) : N=512, K padded 136->160
  v8f acc0 = {}, acc1 = {};
  for (int kk = 0; kk < 5; ++kk) {
    __syncthreads();
    {
      const uint4* src = (const uint4*)(wihb + kk * 16384);
      uint4* dst = (uint4*)s_bk;
#pragma unroll
      for (int u = 0; u < 4; ++u) dst[tid + u * 512] = src[tid + u * 512];
    }
    __syncthreads();
    int kbase = kk * 32 + khalf;
    const float4* zp = (const float4*)(s_z + row * 160 + kbase);
    float4 z0 = zp[0], z1 = zp[1], z2 = zp[2], z3 = zp[3];
    v16bf a;
    a[0]=(__bf16)z0.x; a[1]=(__bf16)z0.y; a[2]=(__bf16)z0.z; a[3]=(__bf16)z0.w;
    a[4]=(__bf16)z1.x; a[5]=(__bf16)z1.y; a[6]=(__bf16)z1.z; a[7]=(__bf16)z1.w;
    a[8]=(__bf16)z2.x; a[9]=(__bf16)z2.y; a[10]=(__bf16)z2.z; a[11]=(__bf16)z2.w;
    a[12]=(__bf16)z3.x; a[13]=(__bf16)z3.y; a[14]=(__bf16)z3.z; a[15]=(__bf16)z3.w;
    v16bf bv0 = *(const v16bf*)(s_bk + (wave * 16 + row) * 32 + khalf);
    v16bf bv1 = *(const v16bf*)(s_bk + ((wave + 16) * 16 + row) * 32 + khalf);
    acc0 = wmma_bf16(a, bv0, acc0);
    acc1 = wmma_bf16(a, bv1, acc1);
  }
  {
    int n0 = wave * 16 + row, n1 = (wave + 16) * 16 + row;
    float bias0 = bih[n0] + bhh[n0];
    float bias1 = bih[n1] + bhh[n1];
#pragma unroll
    for (int r = 0; r < 8; ++r) {
      int m = r + mofs;
      xp[(size_t)(b0 + m) * GDIM + n0] = acc0[r] + bias0;
      xp[(size_t)(b0 + m) * GDIM + n1] = acc1[r] + bias1;
    }
  }
}

// ===================== LSTM (serial, latency-optimized) =====================
__global__ __launch_bounds__(512) void lstm_kernel(
    const float* __restrict__ xp, const float* __restrict__ whh,
    __bf16* __restrict__ hs, float* __restrict__ out)
{
  __shared__ __align__(16) float s_h[HDIM];
  __shared__ float s_g[GDIM];
  const int tid = threadIdx.x;

  float4 wr[32];
  const float4* wsrc = (const float4*)(whh + (size_t)tid * HDIM);
#pragma unroll
  for (int j = 0; j < 32; ++j) wr[j] = wsrc[j];

  if (tid < HDIM) s_h[tid] = 0.f;
  float c = 0.f;
  __syncthreads();

  float xv = xp[tid];
  for (int t = 0; t < BATCH; ++t) {
    if (t + 1 < BATCH) __builtin_prefetch(&xp[(size_t)(t + 1) * GDIM + tid], 0, 0);
    float g = xv;
    const float4* h4 = (const float4*)s_h;
#pragma unroll
    for (int j = 0; j < 32; ++j) {
      float4 h = h4[j];
      g += wr[j].x * h.x + wr[j].y * h.y + wr[j].z * h.z + wr[j].w * h.w;
    }
    s_g[tid] = g;
    if (t + 1 < BATCH) xv = xp[(size_t)(t + 1) * GDIM + tid];
    __syncthreads();
    if (tid < HDIM) {
      float ig = sigmoid_f(s_g[tid]);
      float fg = sigmoid_f(s_g[HDIM + tid]);
      float gg = tanhf(s_g[2 * HDIM + tid]);
      float og = sigmoid_f(s_g[3 * HDIM + tid]);
      c = fg * c + ig * gg;
      float h = og * tanhf(c);
      s_h[tid] = h;
      hs[(size_t)t * HDIM + tid] = (__bf16)h;
    }
    __syncthreads();
  }
  if (tid < HDIM) {
    out[(size_t)BATCH * 75 + tid] = s_h[tid];
    out[(size_t)BATCH * 75 + HDIM + tid] = c;
  }
}

// ===================== decoder =====================
#define DEC_OFF_D   0        // bf16 [49][16][64] padded dec-linear out planes (100352 B)
#define DEC_OFF_D2  100352   // bf16 [16][32][49] padded deconv1 out           (50176 B)
#define DEC_OFF_WD1 150528   // bf16 [32][576] effective deconv1 W (TDM-filled) (36864 B)
#define DEC_OFF_HS  187392   // bf16 [16][128]                                 (4096 B)
#define DEC_OFF_W2F 191488   // f32  [3][32][9] effective deconv2 W            (3456 B)
#define DEC_LDS     194944

__global__ __launch_bounds__(512) void decoder_kernel(
    const __bf16* __restrict__ hsb, const __bf16* __restrict__ dwb,
    const float* __restrict__ db,
    const __bf16* __restrict__ t1wb, const float* __restrict__ t1b,
    const float* __restrict__ t2w, const float* __restrict__ t2b,
    float* __restrict__ out)
{
  extern __shared__ char smem[];
  __bf16* s_dp  = (__bf16*)(smem + DEC_OFF_D);    // [plane(49)][b(16)][ci(64)]
  __bf16* s_d2  = (__bf16*)(smem + DEC_OFF_D2);   // [b][32][49] padded
  __bf16* s_wd1 = (__bf16*)(smem + DEC_OFF_WD1);  // [o(32)][k'(576)]
  __bf16* s_hs  = (__bf16*)(smem + DEC_OFF_HS);
  float*  s_w2  = (float*) (smem + DEC_OFF_W2F);

  const int tid = threadIdx.x, lane = tid & 31, wave = tid >> 5;
  const int b0 = blockIdx.x * TILE_B;
  const int row = lane & 15;
  const int khalf = (lane & 16) ? 16 : 0;
  const int mofs  = (lane & 16) ? 8 : 0;

  // deconv1 weights stream in via TDM while we zero + run dec-linear
  if (wave == 0) tdm_load_1d(lds_addr_of(s_wd1), t1wb, 36864);

  { // zero both padded activation buffers (contiguous 150528 B)
    unsigned* z = (unsigned*)s_dp;
    for (int i = tid; i < 37632; i += 512) z[i] = 0u;
  }
  if (tid < 256) ((uint4*)s_hs)[tid] = ((const uint4*)(hsb + (size_t)b0 * HDIM))[tid];
  for (int idx = tid; idx < 3 * 32 * 9; idx += 512) {
    int o = idx / 288, rem = idx % 288, i = rem / 9, rr = rem % 9, dy = rr / 3, dx = rr % 3;
    s_w2[idx] = t2w[i * 27 + o * 9 + (2 - dy) * 3 + (2 - dx)];
  }
  __syncthreads();
  // dec linear via WMMA: M=16, N=1600 (100 tiles), K=128; B frags direct from global bf16
  for (int nt = wave; nt < 100; nt += 16) {
    int n = nt * 16 + row;
    v8f acc = {};
#pragma unroll
    for (int kk = 0; kk < 4; ++kk) {
      int kbase = kk * 32 + khalf;
      v16bf a = *(const v16bf*)(s_hs + row * 128 + kbase);
      v16bf b = *(const v16bf*)(dwb + (size_t)n * 128 + kbase);   // [n][k] bf16
      acc = wmma_bf16(a, b, acc);
    }
    int ch = n / 25, p = n % 25, py = p / 5, px = p % 5;
    float bias = db[n];
#pragma unroll
    for (int r = 0; r < 8; ++r)
      s_dp[((py + 1) * 7 + (px + 1)) * 1024 + (r + mofs) * 64 + ch] = (__bf16)(acc[r] + bias);
  }
  if (wave == 0) __builtin_amdgcn_s_wait_tensorcnt(0);   // deconv1 weights landed
  __syncthreads();
  // deconv1 via WMMA implicit GEMM: n-tile fixed per wave, positions strided (p += 8)
  {
    const int nt = wave & 1, pg = wave >> 1;
    const int n = nt * 16 + row;
    const __bf16* wrow = s_wd1 + n * 576 + khalf;
    const float bias = t1b[n];
    for (int p = pg; p < 25; p += 8) {
      int py = p / 5, px = p % 5;
      const __bf16* abase = s_dp + (py * 7 + px) * 1024 + row * 64 + khalf;
      v8f acc = {};
#pragma unroll
      for (int kk = 0; kk < 18; ++kk) {
        const int e = kk >> 1, chalf = (kk & 1) * 32;
        const int dy = e / 3, dx = e % 3;
        v16bf a = *(const v16bf*)(abase + (dy * 7 + dx) * 1024 + chalf);
        v16bf b = *(const v16bf*)(wrow + kk * 32);
        acc = wmma_bf16(a, b, acc);
      }
#pragma unroll
      for (int r = 0; r < 8; ++r)
        s_d2[((r + mofs) * 32 + n) * 49 + (py + 1) * 7 + (px + 1)] =
            (__bf16)fmaxf(acc[r] + bias, 0.f);
    }
  }
  __syncthreads();
  // deconv2 (32->3, small) + sigmoid on VALU
  for (int idx = tid; idx < TILE_B * 75; idx += 512) {
    int b = idx / 75, rem = idx % 75, o = rem / 25, p = rem % 25;
    int py = p / 5, px = p % 5;
    float acc = t2b[o];
#pragma unroll
    for (int ci = 0; ci < 32; ++ci)
#pragma unroll
      for (int dy = 0; dy < 3; ++dy)
#pragma unroll
        for (int dx = 0; dx < 3; ++dx)
          acc += (float)s_d2[(b * 32 + ci) * 49 + (py + dy) * 7 + (px + dx)] *
                 s_w2[o * 288 + ci * 9 + dy * 3 + dx];
    out[(size_t)(b0 + b) * 75 + rem] = 1.f / (1.f + __expf(-acc));
  }
}

// ===================== launch =====================
extern "C" void kernel_launch(void* const* d_in, const int* in_sizes, int n_in,
                              void* d_out, int out_size, void* d_ws, size_t ws_size,
                              hipStream_t stream) {
  (void)in_sizes; (void)n_in; (void)out_size; (void)ws_size;
  const float* img = (const float*)d_in[0];
  const float* action = (const float*)d_in[1];
  const float* c1w = (const float*)d_in[2];
  const float* c1b = (const float*)d_in[3];
  const float* c2w = (const float*)d_in[4];
  const float* c2b = (const float*)d_in[5];
  const float* ew  = (const float*)d_in[6];
  const float* eb  = (const float*)d_in[7];
  const float* wih = (const float*)d_in[8];
  const float* whh = (const float*)d_in[9];
  const float* bih = (const float*)d_in[10];
  const float* bhh = (const float*)d_in[11];
  const float* dw  = (const float*)d_in[12];
  const float* db  = (const float*)d_in[13];
  const float* t1w = (const float*)d_in[14];
  const float* t1b = (const float*)d_in[15];
  const float* t2w = (const float*)d_in[16];
  const float* t2b = (const float*)d_in[17];
  float* out = (float*)d_out;

  char* ws = (char*)d_ws;
  float*  xp   = (float*) (ws + WS_XP);
  __bf16* hsb  = (__bf16*)(ws + WS_HSB);
  __bf16* ewb  = (__bf16*)(ws + WS_EWB);
  __bf16* wihb = (__bf16*)(ws + WS_WIHB);
  __bf16* c2wb = (__bf16*)(ws + WS_C2WB);
  __bf16* t1wb = (__bf16*)(ws + WS_T1WB);
  __bf16* dwb  = (__bf16*)(ws + WS_DWB);

  prep_weights_kernel<<<800, 256, 0, stream>>>(ew, wih, c2w, t1w, dw,
                                               ewb, wihb, c2wb, t1wb, dwb);
  encoder_kernel<<<BATCH / TILE_B, 512, ENC_LDS, stream>>>(
      img, action, c1w, c1b, c2wb, c2b, ewb, eb, wihb, bih, bhh, xp);
  lstm_kernel<<<1, 512, 0, stream>>>(xp, whh, hsb, out);
  decoder_kernel<<<BATCH / TILE_B, 512, DEC_LDS, stream>>>(
      hsb, dwb, db, t1wb, t1b, t2w, t2b, out);
}
